// SelfAttention_6296422056616
// MI455X (gfx1250) — matrix operand order repaired and checked by reading, not compiled, not run on hardware
//
#include <hip/hip_runtime.h>

// ---------------------------------------------------------------------------
// CDNA5 (gfx1250) fused self-attention block, wave32 + v_wmma_f32_16x16x32_bf16
// ---------------------------------------------------------------------------

typedef __attribute__((ext_vector_type(16))) __bf16 v16bf;
typedef __attribute__((ext_vector_type(8)))  __bf16 v8bf;
typedef __attribute__((ext_vector_type(8)))  float  v8f;

#define BB 16
#define CC 256
#define HH 64
#define WW 64
#define NN 4096      // H*W
#define NP 1024      // pooled N/4
#define DQK 32       // C/8
#define DV  128      // C/2
#define PROJ_ROWS 192 // 32 theta + 32 phi + 128 g

// ---- WMMA helper ----------------------------------------------------------
__device__ __forceinline__ v8f wmma_bf16(v16bf a, v16bf b, v8f c) {
  // D = A(16x32 bf16) * B(32x16 bf16) + C(16x16 f32)
  return __builtin_amdgcn_wmma_f32_16x16x32_bf16(
      /*neg_a=*/false, a, /*neg_b=*/false, b,
      /*c_mod=*/(short)0, c, /*reuse_a=*/false, /*reuse_b=*/false);
}

// A fragment (16x32, M x K): matrix row-major [row][k], leading dim ld (halves).
// ISA layout: lane m (0-15) holds K=0..7 and K=16..23; lane m+16 holds K=8..15, 24..31.
__device__ __forceinline__ v16bf fragA(const __bf16* m, int ld, int row, int k0, int lane) {
  const __bf16* p = m + (size_t)(row + (lane & 15)) * ld + k0 + ((lane >> 4) & 1) * 8;
  union { v16bf v; v8bf h[2]; } u;
  u.h[0] = *(const v8bf*)(p);
  u.h[1] = *(const v8bf*)(p + 16);
  return u.v;
}

// B fragment (32x16, K x N): stored col-major-as-rows [col][k], leading dim ld.
// ISA layout: lane n (0-15) holds K=0..7 and K=16..23 of column n; lane n+16 holds K=8..15, 24..31.
__device__ __forceinline__ v16bf fragB(const __bf16* m, int ld, int col, int k0, int lane) {
  const __bf16* p = m + (size_t)(col + (lane & 15)) * ld + k0 + ((lane >> 4) & 1) * 8;
  union { v16bf v; v8bf h[2]; } u;
  u.h[0] = *(const v8bf*)(p);
  u.h[1] = *(const v8bf*)(p + 16);
  return u.v;
}

// ---- 1) pack weights to bf16 ---------------------------------------------
__global__ void pack_weights_k(const float* __restrict__ wt, const float* __restrict__ wp,
                               const float* __restrict__ wg, const float* __restrict__ wf,
                               __bf16* __restrict__ wAll, __bf16* __restrict__ wfb) {
  int i = blockIdx.x * 256 + threadIdx.x;
  if (i < PROJ_ROWS * CC) {
    int o = i >> 8;
    float v = (o < 32) ? wt[i] : (o < 64) ? wp[i - 32 * CC] : wg[i - 64 * CC];
    wAll[i] = (__bf16)v;
  }
  if (i < CC * DV) wfb[i] = (__bf16)wf[i];
}

// ---- 2) transpose-convert x: [B][C][N] f32 -> [B][N][C] bf16 --------------
__global__ void transpose_x_k(const float* __restrict__ x, __bf16* __restrict__ xT) {
  __shared__ float tile[32][33];
  int b = blockIdx.z;
  int c0 = blockIdx.y * 32, n0 = blockIdx.x * 32;
  const float* xb = x + (size_t)b * CC * NN;
  #pragma unroll
  for (int i = 0; i < 32; i += 8)
    tile[threadIdx.y + i][threadIdx.x] =
        xb[(size_t)(c0 + threadIdx.y + i) * NN + n0 + threadIdx.x];
  __syncthreads();
  __bf16* xo = xT + (size_t)b * NN * CC;
  #pragma unroll
  for (int i = 0; i < 32; i += 8)
    xo[(size_t)(n0 + threadIdx.y + i) * CC + c0 + threadIdx.x] =
        (__bf16)tile[threadIdx.x][threadIdx.y + i];
}

// ---- 3) projection GEMM: P[b] = W_all[192,256] @ xT[b]^T -> [192][4096] f32
__global__ void __launch_bounds__(256) proj_gemm_k(
    const __bf16* __restrict__ wAll, const __bf16* __restrict__ xT,
    const float* __restrict__ bTheta, const float* __restrict__ bPhi,
    const float* __restrict__ bG, float* __restrict__ P) {
  int lane = threadIdx.x & 31, wv = threadIdx.x >> 5;
  int tile = blockIdx.x * 8 + wv;              // 16 * 12 * 256 tiles
  int nt = tile & 255; tile >>= 8;
  int ot = tile % 12;  int b = tile / 12;
  int o0 = ot * 16, n0 = nt * 16;
  int hi8 = (lane >= 16) ? 8 : 0;

  v8f acc;                                     // C init = bias (row-striped layout)
  #pragma unroll
  for (int v = 0; v < 8; ++v) {
    int o = o0 + v + hi8;
    acc[v] = (o < 32) ? bTheta[o] : (o < 64) ? bPhi[o - 32] : bG[o - 64];
  }
  const __bf16* xb = xT + (size_t)b * NN * CC;
  #pragma unroll
  for (int kb = 0; kb < CC; kb += 32) {
    v16bf a = fragA(wAll, CC, o0, kb, lane);
    v16bf bb = fragB(xb, CC, n0, kb, lane);
    acc = wmma_bf16(a, bb, acc);
  }
  float* Pb = P + (size_t)b * PROJ_ROWS * NN;
  int cc = lane & 15;
  #pragma unroll
  for (int v = 0; v < 8; ++v)
    Pb[(size_t)(o0 + v + hi8) * NN + n0 + cc] = acc[v];
}

// ---- 4a) theta pack: thetaT[b][n][32] = P rows 0..31 ----------------------
__global__ void theta_pack_k(const float* __restrict__ P, __bf16* __restrict__ thetaT) {
  int i = blockIdx.x * 256 + threadIdx.x;      // 16*4096*32
  int o = i & 31, n = (i >> 5) & (NN - 1), b = i >> 17;
  thetaT[i] = (__bf16)P[((size_t)b * PROJ_ROWS + o) * NN + n];
}

// ---- 4b) phi 2x2 maxpool: phi_pT[b][m][32] from P rows 32..63 -------------
__global__ void phi_pool_k(const float* __restrict__ P, __bf16* __restrict__ phiT) {
  int i = blockIdx.x * 256 + threadIdx.x;      // 16*1024*32
  int o = i & 31, m = (i >> 5) & (NP - 1), b = i >> 15;
  int h2 = m >> 5, w2 = m & 31;
  const float* p = P + ((size_t)b * PROJ_ROWS + 32 + o) * NN + h2 * 128 + w2 * 2;
  float v = fmaxf(fmaxf(p[0], p[1]), fmaxf(p[64], p[65]));
  phiT[i] = (__bf16)v;
}

// ---- 4c) g 2x2 maxpool: g_pT[b][c][m] from P rows 64..191 -----------------
__global__ void g_pool_k(const float* __restrict__ P, __bf16* __restrict__ gT) {
  int i = blockIdx.x * 256 + threadIdx.x;      // 16*128*1024
  int m = i & (NP - 1), c = (i >> 10) & 127, b = i >> 17;
  int h2 = m >> 5, w2 = m & 31;
  const float* p = P + ((size_t)b * PROJ_ROWS + 64 + c) * NN + h2 * 128 + w2 * 2;
  float v = fmaxf(fmaxf(p[0], p[1]), fmaxf(p[64], p[65]));
  gT[i] = (__bf16)v;
}

// ---- 5) fused flash-attention: softmax(theta @ phi) @ g -> out bf16 -------
__global__ void __launch_bounds__(256) attn_k(
    const __bf16* __restrict__ thetaT, const __bf16* __restrict__ phiT,
    const __bf16* __restrict__ gT, __bf16* __restrict__ outA) {
  __shared__ __align__(16) __bf16 pbuf[8][16][40];  // per-wave P-tile transpose buffer
  int lane = threadIdx.x & 31, wv = threadIdx.x >> 5;
  int tile = blockIdx.x * 8 + wv;              // 16 * 256 tiles
  int nt = tile & 255, b = tile >> 8;
  int n0 = nt * 16;
  int hi8 = (lane >= 16) ? 8 : 0, cc = lane & 15;

  const __bf16* th = thetaT + (size_t)b * NN * DQK;
  const __bf16* ph = phiT + (size_t)b * NP * DQK;
  const __bf16* gg = gT + (size_t)b * DV * NP;

  v16bf aTh = fragA(th, DQK, n0, 0, lane);     // resident query tile, K=32 = full d

  float mrow[8], lrow[8];
  v8f acc[8];                                  // 16 x 128 output accumulator
  #pragma unroll
  for (int v = 0; v < 8; ++v) { mrow[v] = -3.0e38f; lrow[v] = 0.f; }
  #pragma unroll
  for (int t = 0; t < 8; ++t)
    #pragma unroll
    for (int v = 0; v < 8; ++v) acc[t][v] = 0.f;

  for (int j = 0; j < NP; j += 32) {
    if (j + 32 < NP)
      __builtin_prefetch(ph + (size_t)(j + 32 + cc) * DQK, 0, 1);

    // logits for 32 kv columns: exactly two K=32 WMMAs
    v8f z; 
    #pragma unroll
    for (int v = 0; v < 8; ++v) z[v] = 0.f;
    v8f S0 = wmma_bf16(aTh, fragB(ph, DQK, j, 0, lane), z);
    v8f S1 = wmma_bf16(aTh, fragB(ph, DQK, j + 16, 0, lane), z);

    // online softmax; row r lives across one 16-lane half in VGPR r%8
    float sc[8];
    #pragma unroll
    for (int v = 0; v < 8; ++v) {
      float t = fmaxf(S0[v], S1[v]);
      t = fmaxf(t, __shfl_xor(t, 1, 32));
      t = fmaxf(t, __shfl_xor(t, 2, 32));
      t = fmaxf(t, __shfl_xor(t, 4, 32));
      t = fmaxf(t, __shfl_xor(t, 8, 32));
      float mn = fmaxf(mrow[v], t);
      sc[v] = __expf(mrow[v] - mn);
      mrow[v] = mn;
      float p0 = __expf(S0[v] - mn);
      float p1 = __expf(S1[v] - mn);
      S0[v] = p0; S1[v] = p1;
      float s = p0 + p1;
      s += __shfl_xor(s, 1, 32);
      s += __shfl_xor(s, 2, 32);
      s += __shfl_xor(s, 4, 32);
      s += __shfl_xor(s, 8, 32);
      lrow[v] = lrow[v] * sc[v] + s;
    }
    #pragma unroll
    for (int t = 0; t < 8; ++t)
      #pragma unroll
      for (int v = 0; v < 8; ++v) acc[t][v] *= sc[v];

    // transpose P (C-layout) -> A-fragment layout through per-wave LDS
    #pragma unroll
    for (int v = 0; v < 8; ++v) {
      pbuf[wv][v + hi8][cc]      = (__bf16)S0[v];
      pbuf[wv][v + hi8][cc + 16] = (__bf16)S1[v];
    }
    // DS is in-order per wave; fence compiler + drain DS pipe
    asm volatile("s_wait_dscnt 0" ::: "memory");
    v16bf aP = fragA(&pbuf[wv][0][0], 40, 0, 0, lane);
    asm volatile("s_wait_dscnt 0" ::: "memory");

    // out += P(16x32) @ g(32kv x 16ch) for each of the 8 channel tiles
    #pragma unroll
    for (int t = 0; t < 8; ++t)
      acc[t] = wmma_bf16(aP, fragB(gg, NP, t * 16, j, lane), acc[t]);
  }

  // normalize and store out[b][n][128] bf16
  #pragma unroll
  for (int v = 0; v < 8; ++v) {
    float inv = 1.0f / lrow[v];
    size_t row = (size_t)(b * NN + n0 + v + hi8) * DV;
    #pragma unroll
    for (int t = 0; t < 8; ++t)
      outA[row + t * 16 + cc] = (__bf16)(acc[t][v] * inv);
  }
}

// ---- 6) final GEMM + residual: y = gamma*(w_f @ out^T + b_f) + x ----------
__global__ void __launch_bounds__(256) final_gemm_k(
    const __bf16* __restrict__ wfb, const __bf16* __restrict__ outA,
    const float* __restrict__ bF, const float* __restrict__ x,
    const float* __restrict__ gamma, float* __restrict__ y) {
  int lane = threadIdx.x & 31, wv = threadIdx.x >> 5;
  int tile = blockIdx.x * 8 + wv;              // 16 * 16 * 256 tiles
  int nt = tile & 255; tile >>= 8;
  int ot = tile & 15;  int b = tile >> 4;
  int o0 = ot * 16, n0 = nt * 16;
  int hi8 = (lane >= 16) ? 8 : 0, cc = lane & 15;

  v8f acc;
  #pragma unroll
  for (int v = 0; v < 8; ++v) acc[v] = bF[o0 + v + hi8];

  const __bf16* ob = outA + (size_t)b * NN * DV;
  #pragma unroll
  for (int kb = 0; kb < DV; kb += 32)
    acc = wmma_bf16(fragA(wfb, DV, o0, kb, lane), fragB(ob, DV, n0, kb, lane), acc);

  float gm = gamma[0];
  #pragma unroll
  for (int v = 0; v < 8; ++v) {
    size_t idx = (size_t)(b * CC + o0 + v + hi8) * NN + n0 + cc;
    y[idx] = gm * acc[v] + x[idx];
  }
}

// ---------------------------------------------------------------------------
extern "C" void kernel_launch(void* const* d_in, const int* in_sizes, int n_in,
                              void* d_out, int out_size, void* d_ws, size_t ws_size,
                              hipStream_t stream) {
  const float* x      = (const float*)d_in[0];
  const float* wTheta = (const float*)d_in[1];
  const float* bTheta = (const float*)d_in[2];
  const float* wPhi   = (const float*)d_in[3];
  const float* bPhi   = (const float*)d_in[4];
  const float* wG     = (const float*)d_in[5];
  const float* bG     = (const float*)d_in[6];
  const float* wF     = (const float*)d_in[7];
  const float* bF     = (const float*)d_in[8];
  const float* gamma  = (const float*)d_in[9];
  float* y = (float*)d_out;

  // carve workspace (all offsets 256B aligned)
  char* w = (char*)d_ws;
  auto carve = [&](size_t bytes) {
    void* p = (void*)w;
    w += (bytes + 255) & ~(size_t)255;
    return p;
  };
  __bf16* xT     = (__bf16*)carve((size_t)BB * NN * CC * 2);        // 32 MiB
  __bf16* wAll   = (__bf16*)carve((size_t)PROJ_ROWS * CC * 2);
  __bf16* wfb    = (__bf16*)carve((size_t)CC * DV * 2);
  float*  P      = (float*) carve((size_t)BB * PROJ_ROWS * NN * 4); // 48 MiB
  __bf16* thetaT = (__bf16*)carve((size_t)BB * NN * DQK * 2);
  __bf16* phiT   = (__bf16*)carve((size_t)BB * NP * DQK * 2);
  __bf16* gT     = (__bf16*)carve((size_t)BB * DV * NP * 2);
  __bf16* outA   = (__bf16*)carve((size_t)BB * NN * DV * 2);        // 16 MiB

  // 1) weights -> bf16
  pack_weights_k<<<(PROJ_ROWS * CC + 255) / 256, 256, 0, stream>>>(
      wTheta, wPhi, wG, wF, wAll, wfb);
  // 2) x -> xT bf16
  transpose_x_k<<<dim3(NN / 32, CC / 32, BB), dim3(32, 8), 0, stream>>>(x, xT);
  // 3) fused projection GEMM (theta|phi|g), bias folded in
  proj_gemm_k<<<(BB * 12 * 256) / 8, 256, 0, stream>>>(wAll, xT, bTheta, bPhi, bG, P);
  // 4) pack / pool to WMMA layouts
  theta_pack_k<<<(BB * NN * DQK) / 256, 256, 0, stream>>>(P, thetaT);
  phi_pool_k<<<(BB * NP * DQK) / 256, 256, 0, stream>>>(P, phiT);
  g_pool_k<<<(BB * DV * NP) / 256, 256, 0, stream>>>(P, gT);
  // 5) fused flash attention
  attn_k<<<(BB * 256) / 8, 256, 0, stream>>>(thetaT, phiT, gT, outA);
  // 6) final projection + gamma residual
  final_gemm_k<<<(BB * 16 * 256) / 8, 256, 0, stream>>>(wfb, outA, bF, x, gamma, y);
}
